// SpatialTemporalTokenMerger_13134009991728
// MI455X (gfx1250) — compile-verified, matrix operand order checked by reading, not run on hardware
//
#include <hip/hip_runtime.h>
#include <math.h>

// ---- problem constants (match reference) ----
#define Bn   2
#define Tn   32
#define Nn   576
#define Dn   1024
#define HDn  128
#define Sn   8          // round(T*0.25)
#define Cn   288        // round(N*0.5)
#define SEQ  18496      // VE + 48
#define VSv  16
#define Mv   11520      // (Sn+Tn)*Cn
#define EPSf 1e-6f
#define TAUf 0.8f

#define KSTEP 64
#define PITCH 72        // ushorts per LDS tile row (64 data + 8 pad) -> 36-bank stride

typedef __bf16 v16bf __attribute__((ext_vector_type(16)));
typedef float  v8f   __attribute__((ext_vector_type(8)));

// keep the 5 smallest values seen so far in t[5]
__device__ __forceinline__ void ins5(float v, float t[5]) {
    int mi = 0; float mv = t[0];
#pragma unroll
    for (int q = 1; q < 5; ++q) if (t[q] > mv) { mv = t[q]; mi = q; }
    if (v < mv) t[mi] = v;
}

// LDS byte offset of a generic pointer into __shared__ (addr[31:0] is the LDS offset)
__device__ __forceinline__ unsigned lds_off(const void* p) {
    return (unsigned)(size_t)p;
}
// async copy: 16 bytes per lane, global -> LDS, tracked by ASYNCcnt
__device__ __forceinline__ void async_b128(unsigned ldsByteOff, const void* gsrc) {
    asm volatile("global_load_async_to_lds_b128 %0, %1, off"
                 :: "v"(ldsByteOff), "v"(gsrc) : "memory");
}
__device__ __forceinline__ void wait_async0() {
    asm volatile("s_wait_asynccnt 0x0" ::: "memory");
}

// ---------------- Stage A: frame_feat[b,t,d] = mean_n tokens ----------------
__global__ void frame_mean_kernel(const float* __restrict__ hid, float* __restrict__ ff) {
    int bt = blockIdx.x; int b = bt / Tn, t = bt % Tn;
    const float* base = hid + ((size_t)b * SEQ + VSv + (size_t)t * Nn) * Dn;
    for (int d = threadIdx.x; d < Dn; d += blockDim.x) {
        float s = 0.f;
        for (int n = 0; n < Nn; ++n) s += base[(size_t)n * Dn + d];
        ff[(size_t)bt * Dn + d] = s * (1.0f / Nn);
    }
}

// ------- Stage B: tiny DPC-KNN on 32 frame features (one block / batch) -------
__global__ void small_dpc_kernel(const float* __restrict__ ff, int* __restrict__ labT,
                                 float* __restrict__ cntS, float* __restrict__ segT) {
    __shared__ float d2s[Tn][Tn + 1];
    __shared__ float rhoS[Tn], auxS[Tn], gamS[Tn];
    __shared__ int   cenS[Sn];
    __shared__ float dmaxS;
    int b = blockIdx.x, tid = threadIdx.x;
    const float* fb = ff + (size_t)b * Tn * Dn;
    for (int pr = tid; pr < Tn * Tn; pr += blockDim.x) {
        int i = pr / Tn, j = pr % Tn;
        float acc = 0.f;
        for (int d = 0; d < Dn; ++d) {
            float df = fb[(size_t)i * Dn + d] - fb[(size_t)j * Dn + d];
            acc = fmaf(df, df, acc);
        }
        d2s[i][j] = acc;
    }
    __syncthreads();
    if (tid < Tn) {
        float t5[5] = {1e30f, 1e30f, 1e30f, 1e30f, 1e30f};
        float mx = 0.f;
        for (int j = 0; j < Tn; ++j) { float v = d2s[tid][j]; ins5(v, t5); mx = fmaxf(mx, v); }
        float s = t5[0] + t5[1] + t5[2] + t5[3] + t5[4];
        rhoS[tid] = expf(-s * 0.2f);
        auxS[tid] = mx;
    }
    __syncthreads();
    if (tid == 0) { float m = 0.f; for (int j = 0; j < Tn; ++j) m = fmaxf(m, auxS[j]); dmaxS = m; }
    __syncthreads();
    if (tid < Tn) {
        float ri = rhoS[tid], best = dmaxS;
        for (int j = 0; j < Tn; ++j) if (rhoS[j] > ri) best = fminf(best, d2s[tid][j]);
        gamS[tid] = ri * best;
    }
    __syncthreads();
    if (tid < Tn) {
        float gi = gamS[tid]; int rank = 0;
        for (int j = 0; j < Tn; ++j) rank += (gamS[j] > gi) || (gamS[j] == gi && j < tid);
        if (rank < Sn) cenS[rank] = tid;
    }
    __syncthreads();
    if (tid < Tn) {
        float best = 1e30f; int bi = 0;
        for (int s = 0; s < Sn; ++s) { float v = d2s[tid][cenS[s]]; if (v < best) { best = v; bi = s; } }
        labT[b * Tn + tid] = bi;
    }
    __syncthreads();
    if (tid < Sn) {
        float cnt = 0.f, ts = 0.f;
        for (int t = 0; t < Tn; ++t)
            if (labT[b * Tn + t] == tid) { cnt += 1.f; ts += (float)t; }
        cntS[b * Sn + tid] = cnt;
        segT[b * Sn + tid] = ts / (cnt + EPSf);
    }
}

// --------- Stage C: segment means of tokens / cos / sin (one block per (b,s,n)) ---------
__global__ void seg_mean_kernel(const float* __restrict__ hid, const float* __restrict__ cosIn,
                                const float* __restrict__ sinIn, const int* __restrict__ labT,
                                const float* __restrict__ cntS, float* __restrict__ segMean,
                                float* __restrict__ cosSeg, float* __restrict__ sinSeg) {
    __shared__ int labS[Tn];
    int blk = blockIdx.x;
    int n = blk % Nn; int s = (blk / Nn) % Sn; int b = blk / (Nn * Sn);
    int tid = threadIdx.x;
    if (tid < Tn) labS[tid] = labT[b * Tn + tid];
    __syncthreads();
    float inv = 1.0f / (cntS[b * Sn + s] + EPSf);
    float acc[7] = {0, 0, 0, 0, 0, 0, 0};
    for (int t = 0; t < Tn; ++t) {
        if (labS[t] != s) continue;
        size_t tokBase = ((size_t)b * SEQ + VSv + (size_t)t * Nn + n) * Dn;
#pragma unroll
        for (int j = 0; j < 7; ++j) {
            int idx = tid + j * 256;
            float v;
            if (idx < Dn) v = hid[tokBase + idx];
            else {
                int q = idx - Dn;          // 0..767
                int which = q / 384;       // 0=cos 1=sin
                int pc = q % 384; int p = pc >> 7, h = pc & 127;
                size_t a = (((size_t)p * Bn + b) * SEQ + VSv + (size_t)t * Nn + n) * HDn + h;
                v = which ? sinIn[a] : cosIn[a];
            }
            acc[j] += v;
        }
    }
    size_t sm = (size_t)(b * Sn + s) * Nn + n;
#pragma unroll
    for (int j = 0; j < 7; ++j) {
        int idx = tid + j * 256;
        float v = acc[j] * inv;
        if (idx < Dn) segMean[sm * Dn + idx] = v;
        else {
            int q = idx - Dn; int which = q / 384; int pc = q % 384;
            if (which == 0) cosSeg[sm * 384 + pc] = v;
            else            sinSeg[sm * 384 + pc] = v;
        }
    }
}

// --------- Stage D1: cosine sim of token vs its segment mean ---------
__global__ void sim_kernel(const float* __restrict__ hid, const float* __restrict__ segMean,
                           const int* __restrict__ labT, float* __restrict__ simv) {
    __shared__ float red[3][128];
    int gid = blockIdx.x;
    int n = gid % Nn; int t = (gid / Nn) % Tn; int b = gid / (Nn * Tn);
    int s = labT[b * Tn + t];
    const float* a = hid + ((size_t)b * SEQ + VSv + (size_t)t * Nn + n) * Dn;
    const float* c = segMean + ((size_t)(b * Sn + s) * Nn + n) * Dn;
    float dot = 0.f, na = 0.f, nb = 0.f;
    for (int d = threadIdx.x; d < Dn; d += 128) {
        float av = a[d], cv = c[d];
        dot = fmaf(av, cv, dot); na = fmaf(av, av, na); nb = fmaf(cv, cv, nb);
    }
    red[0][threadIdx.x] = dot; red[1][threadIdx.x] = na; red[2][threadIdx.x] = nb;
    __syncthreads();
    for (int off = 64; off > 0; off >>= 1) {
        if (threadIdx.x < off)
            for (int q = 0; q < 3; ++q) red[q][threadIdx.x] += red[q][threadIdx.x + off];
        __syncthreads();
    }
    if (threadIdx.x == 0)
        simv[gid] = red[0][0] / ((sqrtf(red[1][0]) + EPSf) * (sqrtf(red[2][0]) + EPSf));
}

// --------- Stage D2: static mask per (b,s,n) and dynamic weight per (b,t,n) ---------
__global__ void static_dyn_kernel(const float* __restrict__ simv, const int* __restrict__ labT,
                                  const float* __restrict__ cntS, float* __restrict__ statW,
                                  float* __restrict__ dynW) {
    int idx = blockIdx.x * blockDim.x + threadIdx.x;
    if (idx >= Bn * Nn) return;
    int b = idx / Nn, n = idx % Nn;
    float sm[Sn];
#pragma unroll
    for (int s = 0; s < Sn; ++s) sm[s] = 0.f;
    for (int t = 0; t < Tn; ++t) sm[labT[b * Tn + t]] += simv[((size_t)(b * Tn + t)) * Nn + n];
    float st[Sn];
#pragma unroll
    for (int s = 0; s < Sn; ++s) {
        float v = sm[s] / (cntS[b * Sn + s] + EPSf);
        st[s] = (v > TAUf) ? 1.f : 0.f;
        statW[((size_t)(b * Sn + s)) * Nn + n] = st[s];
    }
    for (int t = 0; t < Tn; ++t)
        dynW[((size_t)(b * Tn + t)) * Nn + n] = 1.f - st[labT[b * Tn + t]];
}

// --------- per-group: one-shot f32 -> bf16 conversion (8 elems / thread) ---------
__global__ void cvt_bf16_kernel(const float* __restrict__ X, unsigned short* __restrict__ Xb) {
    int idx = blockIdx.x * blockDim.x + threadIdx.x;
    size_t base = (size_t)idx * 8;
    const float4* p = (const float4*)(X + base);
    float4 a = p[0], b = p[1];
    union { unsigned short u[8]; uint4 q; } o;
    o.u[0] = (unsigned short)(__float_as_uint(a.x) >> 16);
    o.u[1] = (unsigned short)(__float_as_uint(a.y) >> 16);
    o.u[2] = (unsigned short)(__float_as_uint(a.z) >> 16);
    o.u[3] = (unsigned short)(__float_as_uint(a.w) >> 16);
    o.u[4] = (unsigned short)(__float_as_uint(b.x) >> 16);
    o.u[5] = (unsigned short)(__float_as_uint(b.y) >> 16);
    o.u[6] = (unsigned short)(__float_as_uint(b.z) >> 16);
    o.u[7] = (unsigned short)(__float_as_uint(b.w) >> 16);
    *(uint4*)(Xb + base) = o.q;
}

// --------- per-group: squared row norms (from f32 source) ---------
__global__ void x2_kernel(const float* __restrict__ X, float* __restrict__ x2) {
    __shared__ float red[256];
    int i = blockIdx.x;
    float s = 0.f;
    for (int d = threadIdx.x; d < Dn; d += 256) { float v = X[(size_t)i * Dn + d]; s = fmaf(v, v, s); }
    red[threadIdx.x] = s; __syncthreads();
    for (int off = 128; off > 0; off >>= 1) {
        if (threadIdx.x < off) red[threadIdx.x] += red[threadIdx.x + off];
        __syncthreads();
    }
    if (threadIdx.x == 0) x2[i] = red[0];
}

// --------- per-group: WMMA bf16 Gram -> d2 = max(x2_i + x2_j - 2*G, 0) ---------
// 64x64 block tile, 8 waves x two 16x16 accumulators. Double-buffered LDS fed by
// GLOBAL_LOAD_ASYNC_TO_LDS_B128 (ASYNCcnt), K=64 per stage -> 4 WMMA / wave / stage.
__global__ __launch_bounds__(256) void gram_d2_kernel(const unsigned short* __restrict__ Xb,
                                                      const float* __restrict__ x2,
                                                      float* __restrict__ d2) {
    __shared__ unsigned short As[2][64 * PITCH];
    __shared__ unsigned short Bs[2][64 * PITCH];
    const int i0 = blockIdx.y * 64, j0 = blockIdx.x * 64;
    const int tid = threadIdx.x;
    const int wave = tid >> 5, lane = tid & 31;
    const int tm  = wave & 3;          // A tile row within block (0..3)
    const int tn0 = (wave >> 2) * 2;   // first of two B tile cols (0 or 2)
    const int hf  = (lane >> 4) & 1;
    const int ml  = lane & 15;

    // this thread's two (row, k-chunk) slots within a 64x64 bf16 tile (16B chunks)
    const int q0 = tid,      r0 = q0 >> 3, k0c = (q0 & 7) * 8;
    const int q1 = tid + 256, r1 = q1 >> 3, k1c = (q1 & 7) * 8;

    auto stage = [&](int buf, int kk) {
        async_b128(lds_off(&As[buf][r0 * PITCH + k0c]), Xb + (size_t)(i0 + r0) * Dn + kk + k0c);
        async_b128(lds_off(&As[buf][r1 * PITCH + k1c]), Xb + (size_t)(i0 + r1) * Dn + kk + k1c);
        async_b128(lds_off(&Bs[buf][r0 * PITCH + k0c]), Xb + (size_t)(j0 + r0) * Dn + kk + k0c);
        async_b128(lds_off(&Bs[buf][r1 * PITCH + k1c]), Xb + (size_t)(j0 + r1) * Dn + kk + k1c);
    };

    v8f acc0 = {}; v8f acc1 = {};
    stage(0, 0);
    wait_async0();
    __syncthreads();
    for (int kk = 0; kk < Dn; kk += KSTEP) {
        const int cur = (kk / KSTEP) & 1, nxt = cur ^ 1;
        if (kk + KSTEP < Dn) stage(nxt, kk + KSTEP);   // overlap copy with math
#pragma unroll
        for (int kh = 0; kh < 2; ++kh) {
            union { unsigned short u[16]; v16bf v; } af, b0f, b1f;
#pragma unroll
            for (int e = 0; e < 16; ++e) {
                int vg = e >> 1;
                int k = kh * 32 + ((vg < 4) ? (vg * 2) : (16 + (vg - 4) * 2)) + hf * 8 + (e & 1);
                af.u[e]  = As[cur][(tm * 16 + ml) * PITCH + k];
                b0f.u[e] = Bs[cur][(tn0 * 16 + ml) * PITCH + k];
                b1f.u[e] = Bs[cur][((tn0 + 1) * 16 + ml) * PITCH + k];
            }
            acc0 = __builtin_amdgcn_wmma_f32_16x16x32_bf16(false, af.v, false, b0f.v, (short)0, acc0, false, false);
            acc1 = __builtin_amdgcn_wmma_f32_16x16x32_bf16(false, af.v, false, b1f.v, (short)0, acc1, false, false);
        }
        wait_async0();     // my prefetch done; barrier makes everyone's prefetch visible
        __syncthreads();
    }
#pragma unroll
    for (int r = 0; r < 8; ++r) {
        int gi  = i0 + tm * 16 + r + hf * 8;     // C/D layout: VGPR r -> M=r / r+8 per lane half
        int gj0 = j0 + tn0 * 16 + ml;
        int gj1 = j0 + (tn0 + 1) * 16 + ml;
        float xi = x2[gi];
        d2[(size_t)gi * Nn + gj0] = fmaxf(xi + x2[gj0] - 2.0f * acc0[r], 0.0f);
        d2[(size_t)gi * Nn + gj1] = fmaxf(xi + x2[gj1] - 2.0f * acc1[r], 0.0f);
    }
}

// --------- per-group: rho[i] = exp(-mean of 5 smallest d2 in row), row max ---------
__global__ void knn_rho_kernel(const float* __restrict__ d2, float* __restrict__ rho,
                               float* __restrict__ rmax) {
    __shared__ float cand[128 * 5];
    __shared__ float smax[128];
    int i = blockIdx.x, tid = threadIdx.x;
    float t5[5] = {1e30f, 1e30f, 1e30f, 1e30f, 1e30f};
    float mx = 0.f;
    for (int j = tid; j < Nn; j += 128) { float v = d2[(size_t)i * Nn + j]; ins5(v, t5); mx = fmaxf(mx, v); }
#pragma unroll
    for (int q = 0; q < 5; ++q) cand[tid * 5 + q] = t5[q];
    smax[tid] = mx;
    __syncthreads();
    if (tid == 0) {
        float f5[5] = {1e30f, 1e30f, 1e30f, 1e30f, 1e30f};
        for (int z = 0; z < 128 * 5; ++z) ins5(cand[z], f5);
        float s = f5[0] + f5[1] + f5[2] + f5[3] + f5[4];
        rho[i] = expf(-s * 0.2f);
        float m = 0.f;
        for (int z = 0; z < 128; ++z) m = fmaxf(m, smax[z]);
        rmax[i] = m;
    }
}

__global__ void dmax_kernel(const float* __restrict__ rmax, float* __restrict__ dmx) {
    __shared__ float red[256];
    float m = 0.f;
    for (int j = threadIdx.x; j < Nn; j += 256) m = fmaxf(m, rmax[j]);
    red[threadIdx.x] = m; __syncthreads();
    for (int off = 128; off > 0; off >>= 1) {
        if (threadIdx.x < off) red[threadIdx.x] = fmaxf(red[threadIdx.x], red[threadIdx.x + off]);
        __syncthreads();
    }
    if (threadIdx.x == 0) dmx[0] = red[0];
}

// --------- per-group: delta (min d2 to any denser point, capped at dmax), gamma ---------
__global__ void delta_gamma_kernel(const float* __restrict__ d2, const float* __restrict__ rho,
                                   const float* __restrict__ dmx, float* __restrict__ gam) {
    __shared__ float red[128];
    int i = blockIdx.x, tid = threadIdx.x;
    float ri = rho[i];
    float best = 1e30f;
    for (int j = tid; j < Nn; j += 128)
        if (rho[j] > ri) best = fminf(best, d2[(size_t)i * Nn + j]);
    red[tid] = best; __syncthreads();
    for (int off = 64; off > 0; off >>= 1) {
        if (tid < off) red[tid] = fminf(red[tid], red[tid + off]);
        __syncthreads();
    }
    if (tid == 0) gam[i] = ri * fminf(red[0], dmx[0]);
}

// --------- per-group: top-Cn of gamma via stable rank ---------
__global__ void centers_kernel(const float* __restrict__ gam, int* __restrict__ cent) {
    __shared__ float g[Nn];
    int i = threadIdx.x;
    g[i] = gam[i];
    __syncthreads();
    float gi = g[i]; int rank = 0;
    for (int j = 0; j < Nn; ++j) rank += (g[j] > gi) || (g[j] == gi && j < i);
    if (rank < Cn) cent[rank] = i;
}

// --------- per-group: label[i] = argmin_c d2[i, centers[c]] ---------
__global__ void label_kernel(const float* __restrict__ d2, const int* __restrict__ cent,
                             int* __restrict__ labs) {
    __shared__ int cs[Cn];
    for (int c = threadIdx.x; c < Cn; c += blockDim.x) cs[c] = cent[c];
    __syncthreads();
    int i = blockIdx.x * blockDim.x + threadIdx.x;
    if (i >= Nn) return;
    float best = 1e30f; int bi = 0;
    for (int c = 0; c < Cn; ++c) {
        float v = d2[(size_t)i * Nn + cs[c]];
        if (v < best) { best = v; bi = c; }
    }
    labs[i] = bi;
}

// --------- per-group: weighted cluster means -> final outputs (deterministic scan) ---------
__global__ void cluster_out_kernel(const float* __restrict__ vals, const int* __restrict__ labs,
                                   const float* __restrict__ wv, const float* __restrict__ cb,
                                   const float* __restrict__ sb, long nStr, long pStr,
                                   const float* __restrict__ segT, int b, int u,
                                   float* __restrict__ outTok, int* __restrict__ outPos,
                                   float* __restrict__ outCos, float* __restrict__ outSin) {
    __shared__ int   slab[Nn];
    __shared__ float sw[Nn];
    int c = blockIdx.x, tid = threadIdx.x;
    for (int n = tid; n < Nn; n += 256) { slab[n] = labs[n]; sw[n] = wv[n]; }
    __syncthreads();
    float tacc[4] = {0, 0, 0, 0};
    float cacc[2] = {0, 0}, sacc[2] = {0, 0};
    float denom = 0.f, p1 = 0.f, p2 = 0.f;
    for (int n = 0; n < Nn; ++n) {
        if (slab[n] != c) continue;           // uniform branch (LDS broadcast)
        float wn = sw[n];
        if (wn == 0.f) continue;
        denom += wn;
        p1 += wn * (float)(n / 24);
        p2 += wn * (float)(n % 24);
        const float* vr = vals + (size_t)n * Dn;
#pragma unroll
        for (int j = 0; j < 4; ++j) tacc[j] += wn * vr[tid + j * 256];
#pragma unroll
        for (int j = 0; j < 2; ++j) {
            int idx = tid + j * 256;
            if (idx < 384) {
                int p = idx >> 7, h = idx & 127;
                cacc[j] += wn * cb[(size_t)n * nStr + (size_t)p * pStr + h];
                sacc[j] += wn * sb[(size_t)n * nStr + (size_t)p * pStr + h];
            }
        }
    }
    float inv = 1.0f / (denom + EPSf);
#pragma unroll
    for (int j = 0; j < 4; ++j) outTok[(size_t)c * Dn + tid + j * 256] = tacc[j] * inv;
#pragma unroll
    for (int j = 0; j < 2; ++j) {
        int idx = tid + j * 256;
        if (idx < 384) {
            int p = idx >> 7, h = idx & 127;
            size_t off = (size_t)p * ((size_t)Bn * Mv * HDn) + (size_t)c * HDn + h;
            outCos[off] = cacc[j] * inv;
            outSin[off] = sacc[j] * inv;
        }
    }
    if (tid == 0) {
        float p0 = (u < Sn) ? segT[b * Sn + u] : (float)(u - Sn);
        outPos[c]                        = (int)rintf(p0 * denom * inv);
        outPos[(size_t)Bn * Mv + c]      = (int)rintf(p1 * inv);
        outPos[2 * (size_t)Bn * Mv + c]  = (int)rintf(p2 * inv);
    }
}

extern "C" void kernel_launch(void* const* d_in, const int* in_sizes, int n_in,
                              void* d_out, int out_size, void* d_ws, size_t ws_size,
                              hipStream_t stream) {
    (void)in_sizes; (void)n_in; (void)out_size; (void)ws_size;
    const float* hidden = (const float*)d_in[0];
    // d_in[1] = position_ids: visual-span positions are structured (t, n/24, n%24) -> unused
    const float* cosIn = (const float*)d_in[2];
    const float* sinIn = (const float*)d_in[3];
    float* out = (float*)d_out;
    float* ws  = (float*)d_ws;

    // workspace layout (float words)
    size_t o = 0;
    float* ff      = ws + o; o += (size_t)Bn * Tn * Dn;
    float* cntS    = ws + o; o += Bn * Sn;
    float* segT    = ws + o; o += Bn * Sn;
    float* segMean = ws + o; o += (size_t)Bn * Sn * Nn * Dn;
    float* cosSeg  = ws + o; o += (size_t)Bn * Sn * Nn * 384;
    float* sinSeg  = ws + o; o += (size_t)Bn * Sn * Nn * 384;
    float* simv    = ws + o; o += (size_t)Bn * Tn * Nn;
    float* statW   = ws + o; o += (size_t)Bn * Sn * Nn;
    float* dynW    = ws + o; o += (size_t)Bn * Tn * Nn;
    float* x2      = ws + o; o += Nn;
    float* d2      = ws + o; o += (size_t)Nn * Nn;
    float* rho     = ws + o; o += Nn;
    float* rmax    = ws + o; o += Nn;
    float* gam     = ws + o; o += Nn;
    float* dmx     = ws + o; o += 1;
    unsigned short* Xb = (unsigned short*)(ws + o); o += ((size_t)Nn * Dn) / 2;
    int*   labT    = (int*)(ws + o); o += Bn * Tn;
    int*   cent    = (int*)(ws + o); o += Cn;
    int*   labs    = (int*)(ws + o); o += Nn;

    frame_mean_kernel<<<Bn * Tn, 256, 0, stream>>>(hidden, ff);
    small_dpc_kernel<<<Bn, 256, 0, stream>>>(ff, labT, cntS, segT);
    seg_mean_kernel<<<Bn * Sn * Nn, 256, 0, stream>>>(hidden, cosIn, sinIn, labT, cntS,
                                                      segMean, cosSeg, sinSeg);
    sim_kernel<<<Bn * Tn * Nn, 128, 0, stream>>>(hidden, segMean, labT, simv);
    static_dyn_kernel<<<(Bn * Nn + 255) / 256, 256, 0, stream>>>(simv, labT, cntS, statW, dynW);

    const size_t O1 = (size_t)Bn * Mv * Dn;              // merged_pos (int32)
    const size_t O2 = O1 + (size_t)3 * Bn * Mv;          // merged_cos
    const size_t O3 = O2 + (size_t)3 * Bn * Mv * HDn;    // merged_sin

    for (int b = 0; b < Bn; ++b) {
        for (int u = 0; u < Sn + Tn; ++u) {
            const float *X, *wv, *cb, *sb;
            long nStr, pStr;
            if (u < Sn) {                 // static path: cluster segment means
                X  = segMean + (size_t)(b * Sn + u) * Nn * Dn;
                wv = statW   + (size_t)(b * Sn + u) * Nn;
                cb = cosSeg  + (size_t)(b * Sn + u) * Nn * 384;
                sb = sinSeg  + (size_t)(b * Sn + u) * Nn * 384;
                nStr = 384; pStr = 128;
            } else {                      // dynamic path: cluster each frame's tokens
                int t = u - Sn;
                X  = hidden + ((size_t)b * SEQ + VSv + (size_t)t * Nn) * Dn;
                wv = dynW + (size_t)(b * Tn + t) * Nn;
                cb = cosIn + ((size_t)b * SEQ + VSv + (size_t)t * Nn) * HDn;
                sb = sinIn + ((size_t)b * SEQ + VSv + (size_t)t * Nn) * HDn;
                nStr = HDn; pStr = (long)Bn * SEQ * HDn;
            }
            cvt_bf16_kernel<<<(Nn * Dn / 8) / 256, 256, 0, stream>>>(X, Xb);
            x2_kernel<<<Nn, 256, 0, stream>>>(X, x2);
            gram_d2_kernel<<<dim3(9, 9), 256, 0, stream>>>(Xb, x2, d2);
            knn_rho_kernel<<<Nn, 128, 0, stream>>>(d2, rho, rmax);
            dmax_kernel<<<1, 256, 0, stream>>>(rmax, dmx);
            delta_gamma_kernel<<<Nn, 128, 0, stream>>>(d2, rho, dmx, gam);
            centers_kernel<<<1, Nn, 0, stream>>>(gam, cent);
            label_kernel<<<(Nn + 127) / 128, 128, 0, stream>>>(d2, cent, labs);

            float* outTok = out + ((size_t)b * Mv + (size_t)u * Cn) * Dn;
            int*   outPos = (int*)(out + O1) + (size_t)b * Mv + (size_t)u * Cn;
            float* outCos = out + O2 + ((size_t)b * Mv + (size_t)u * Cn) * HDn;
            float* outSin = out + O3 + ((size_t)b * Mv + (size_t)u * Cn) * HDn;
            cluster_out_kernel<<<Cn, 256, 0, stream>>>(X, labs, wv, cb, sb, nStr, pStr,
                                                       segT, b, u, outTok, outPos, outCos, outSin);
        }
    }
}
